// MultiScaleAttention_34677565948800
// MI455X (gfx1250) — matrix operand order repaired and checked by reading, not compiled, not run on hardware
//
#include <hip/hip_runtime.h>

#define BB  2
#define NQL 2048
#define NKL 2048
#define DD  2048
#define HH  16
#define HDI 128

typedef __attribute__((ext_vector_type(16))) __bf16 v16bf;
typedef __attribute__((ext_vector_type(8)))  __bf16 v8bf;
typedef __attribute__((ext_vector_type(8)))  float  v8f;

// ---------------- CDNA5 async copy (guarded) ------------------------------
#if defined(__has_builtin)
#if __has_builtin(__builtin_amdgcn_global_load_async_to_lds_b128) && \
    __has_builtin(__builtin_amdgcn_s_wait_asynccnt)
#define HAS_ASYNC_LDS 1
#endif
#endif
#ifndef HAS_ASYNC_LDS
#define HAS_ASYNC_LDS 0
#endif

#if HAS_ASYNC_LDS
typedef __attribute__((ext_vector_type(4))) int i32x4;
typedef __attribute__((address_space(1))) i32x4 i32x4_g;   // global
typedef __attribute__((address_space(3))) i32x4 i32x4_l;   // LDS
#endif

static __device__ __forceinline__ void cp16_g2l(void* lds_dst, const void* gsrc) {
#if HAS_ASYNC_LDS
  __builtin_amdgcn_global_load_async_to_lds_b128(
      (i32x4_g*)(unsigned long long)(uintptr_t)gsrc,
      (i32x4_l*)(unsigned int)(uintptr_t)lds_dst,
      0, 0);
#else
  *(v8bf*)lds_dst = *(const v8bf*)gsrc;   // sync fallback: load + ds_store
#endif
}

static __device__ __forceinline__ void wait_async_le(int n) {
#if HAS_ASYNC_LDS
  if (n == 0) __builtin_amdgcn_s_wait_asynccnt(0);
  else        __builtin_amdgcn_s_wait_asynccnt(5);
#endif
  (void)n;
}

// ---------------- WMMA helpers -------------------------------------------

static __device__ __forceinline__ v8f wmma_bf16(v16bf a, v16bf b, v8f c) {
  return __builtin_amdgcn_wmma_f32_16x16x32_bf16(false, a, false, b,
                                                 (short)0, c, false, false);
}

// A-matrix 16x32 fragment: per lane two groups of 8 contiguous K elements.
// Also used for contiguous-16 loads by passing p1 = p0 + 8.
static __device__ __forceinline__ v16bf frag2_bf16(const __bf16* p0,
                                                   const __bf16* p1) {
  v8bf lo = *(const v8bf*)p0;
  v8bf hi = *(const v8bf*)p1;
  v16bf r;
#pragma unroll
  for (int i = 0; i < 8; ++i) { r[i] = lo[i]; r[i + 8] = hi[i]; }
  return r;
}

// B-matrix 32x16 fragment: 16 contiguous K elements (32B-aligned global).
static __device__ __forceinline__ v16bf frag_b_bf16(const __bf16* p) {
  return *(const v16bf*)p;
}

// ---------------- prep: fp32 -> bf16 elementwise --------------------------
__global__ __launch_bounds__(256) void conv_bf16_kernel(
    const float* __restrict__ in, __bf16* __restrict__ out) {
  size_t i = (size_t)blockIdx.x * 256 + threadIdx.x;
  out[i] = (__bf16)in[i];
}

// K: fp32 [B,H,NKV,HD] -> bf16 same layout.
// V: fp32 [B,H,NKV,HD] -> bf16 transposed [B,H,HD,NKV].
__global__ __launch_bounds__(256) void kv_prep_kernel(
    const float* __restrict__ k, const float* __restrict__ v,
    __bf16* __restrict__ kb, __bf16* __restrict__ vt) {
  int idx = blockIdx.x * 256 + threadIdx.x;
  kb[idx] = (__bf16)k[idx];
  int hd = idx & (HDI - 1);
  int kv = (idx >> 7) & (NKL - 1);
  int bh = idx >> 18;
  vt[((size_t)bh * HDI + hd) * NKL + kv] = (__bf16)v[idx];
}

// ---------------- GEMM: C[m,n] = sum_k A[m,k]*W[n,k] + bias[n] ------------
// A, W pre-converted bf16. Block = 4 waves -> 32x128 tile, double-buffered
// async LDS staging. mode 0: out bf16 permuted [B,H,NQ,HD]; mode 1: fp32.
__global__ __launch_bounds__(128) void gemm_nt_kernel(
    const __bf16* __restrict__ Ab, const __bf16* __restrict__ Wb,
    const float* __restrict__ bias, void* __restrict__ Cout,
    int K, int mode) {
  __shared__ __bf16 lA[2][32][40];
  __shared__ __bf16 lB[2][128][40];

  const int tid  = threadIdx.x;
  const int lane = tid & 31;
  const int wave = tid >> 5;
  const int r16  = lane & 15;
  const int half = lane >> 4;
  const int kb8  = half * 8;
  const int m0   = blockIdx.x * 32;
  const int n0   = blockIdx.y * 128;

  const int crow = tid >> 2;      // copy row (0..31)
  const int cseg = tid & 3;       // 16B segment within a 64B row chunk

  v8f acc[2][2] = {};

  auto issue_chunk = [&](int buf, int kc) {
    // A tile: 32 rows x 32 bf16 (64B/row) = 128 x 16B segments
    cp16_g2l(&lA[buf][crow][cseg * 8],
             Ab + (size_t)(m0 + crow) * K + kc + cseg * 8);
    // B tile: 128 rows x 32 bf16 = 512 x 16B segments
#pragma unroll
    for (int i = 0; i < 4; ++i) {
      int rowb = i * 32 + crow;
      cp16_g2l(&lB[buf][rowb][cseg * 8],
               Wb + (size_t)(n0 + rowb) * K + kc + cseg * 8);
    }
  };

  const int niter = K / 32;
  issue_chunk(0, 0);
  for (int it = 0; it < niter; ++it) {
    const bool more = (it + 1) < niter;
    if (more) issue_chunk((it + 1) & 1, (it + 1) * 32);
    wait_async_le(more ? 5 : 0);
    __syncthreads();

    const int buf = it & 1;
    v16bf af[2], bf_[2];
#pragma unroll
    for (int mi = 0; mi < 2; ++mi)
      af[mi] = frag2_bf16(&lA[buf][mi * 16 + r16][half * 16],
                          &lA[buf][mi * 16 + r16][half * 16 + 8]);
#pragma unroll
    for (int ni = 0; ni < 2; ++ni) {
      const __bf16* p = &lB[buf][wave * 32 + ni * 16 + r16][half * 16];
      bf_[ni] = frag2_bf16(p, p + 8);
    }
#pragma unroll
    for (int mi = 0; mi < 2; ++mi)
#pragma unroll
      for (int ni = 0; ni < 2; ++ni)
        acc[mi][ni] = wmma_bf16(af[mi], bf_[ni], acc[mi][ni]);

    __syncthreads();   // readers done before buffer reuse
  }

#pragma unroll
  for (int mi = 0; mi < 2; ++mi)
#pragma unroll
    for (int ni = 0; ni < 2; ++ni) {
      int n = n0 + wave * 32 + ni * 16 + r16;
      float bv = bias[n];
#pragma unroll
      for (int r = 0; r < 8; ++r) {
        int m = m0 + mi * 16 + r + 8 * half;
        float val = acc[mi][ni][r] + bv;
        if (mode == 0) {
          int b = m >> 11, nq = m & (NQL - 1);
          int h = n >> 7,  hd = n & (HDI - 1);
          ((__bf16*)Cout)[((size_t)(b * HH + h) * NQL + nq) * HDI + hd] =
              (__bf16)val;
        } else {
          ((float*)Cout)[(size_t)m * DD + n] = val;
        }
      }
    }
}

// ---------------- Flash attention (one wave per 16-query tile) ------------
__global__ __launch_bounds__(128) void attn_kernel(
    const __bf16* __restrict__ qb, const __bf16* __restrict__ kb,
    const __bf16* __restrict__ vt, const int* __restrict__ tier_labels,
    const unsigned char* __restrict__ pad_mask,
    const float* __restrict__ tier_bias, __bf16* __restrict__ ob) {
  __shared__ __bf16 lds_p[4][16][40];

  const int lane = threadIdx.x & 31;
  const int wave = threadIdx.x >> 5;
  const int half = lane >> 4;
  const int r16  = lane & 15;
  const int kb8  = half * 8;
  const int qt   = blockIdx.x * 4 + wave;
  const int h    = blockIdx.y;
  const int b    = blockIdx.z;
  const int q0   = qt * 16;
  const float scale = 0.08838834764831845f;   // HD^-0.5
  const float LOG2E = 1.4426950408889634f;

  const __bf16* qbase = qb + ((size_t)(b * HH + h) * NQL + q0 + r16) * HDI;
  v16bf aq[4];
#pragma unroll
  for (int c = 0; c < 4; ++c)
    aq[c] = frag2_bf16(qbase + 32 * c + half * 16, qbase + 32 * c + half * 16 + 8);

  v8f oacc[8] = {};
  float mrow[8], lrow[8];
#pragma unroll
  for (int r = 0; r < 8; ++r) { mrow[r] = -1e30f; lrow[r] = 0.f; }

  const int kv_end = q0 + 16;
  for (int j0 = 0; j0 < kv_end; j0 += 32) {
    v8f s[2] = {};
#pragma unroll
    for (int ss = 0; ss < 2; ++ss) {
      const __bf16* kbase =
          kb + ((size_t)(b * HH + h) * NKL + j0 + 16 * ss + r16) * HDI;
#pragma unroll
      for (int c = 0; c < 4; ++c) {
        v16bf bk = frag_b_bf16(kbase + 32 * c + half * 16);
        s[ss] = wmma_bf16(aq[c], bk, s[ss]);
      }
    }
    float tb[2]; bool pad[2]; int ncol[2];
#pragma unroll
    for (int ss = 0; ss < 2; ++ss) {
      ncol[ss] = j0 + 16 * ss + r16;
      tb[ss]  = tier_bias[tier_labels[b * NKL + ncol[ss]]];
      pad[ss] = pad_mask[b * NKL + ncol[ss]] != 0;
    }
    float escale[8];
#pragma unroll
    for (int r = 0; r < 8; ++r) {
      int mg = q0 + r + 8 * half;
      float v0 = (pad[0] || ncol[0] > mg) ? -1e30f : s[0][r] * scale + tb[0];
      float v1 = (pad[1] || ncol[1] > mg) ? -1e30f : s[1][r] * scale + tb[1];
      float rmax = fmaxf(v0, v1);
#pragma unroll
      for (int msk = 1; msk < 16; msk <<= 1)
        rmax = fmaxf(rmax, __shfl_xor(rmax, msk, 32));
      float newm = fmaxf(mrow[r], rmax);
      float es = __builtin_exp2f((mrow[r] - newm) * LOG2E);
      v0 = __builtin_exp2f((v0 - newm) * LOG2E);
      v1 = __builtin_exp2f((v1 - newm) * LOG2E);
      float rs = v0 + v1;
#pragma unroll
      for (int msk = 1; msk < 16; msk <<= 1)
        rs += __shfl_xor(rs, msk, 32);
      lrow[r] = lrow[r] * es + rs;
      mrow[r] = newm;
      escale[r] = es;
      s[0][r] = v0; s[1][r] = v1;
    }
#pragma unroll
    for (int t = 0; t < 8; ++t)
#pragma unroll
      for (int r = 0; r < 8; ++r)
        oacc[t][r] *= escale[r];

#pragma unroll
    for (int ss = 0; ss < 2; ++ss)
#pragma unroll
      for (int r = 0; r < 8; ++r)
        lds_p[wave][r + 8 * half][16 * ss + r16] = (__bf16)s[ss][r];
    asm volatile("s_wait_dscnt 0x0" ::: "memory");
    v16bf ap = frag2_bf16(&lds_p[wave][r16][half * 16],
                          &lds_p[wave][r16][half * 16 + 8]);

#pragma unroll
    for (int t = 0; t < 8; ++t) {
      const __bf16* vbase =
          vt + ((size_t)(b * HH + h) * HDI + 16 * t + r16) * NKL + j0 + half * 16;
      v16bf bv = frag_b_bf16(vbase);
      oacc[t] = wmma_bf16(ap, bv, oacc[t]);
    }
  }

#pragma unroll
  for (int r = 0; r < 8; ++r) lrow[r] = 1.f / lrow[r];
#pragma unroll
  for (int t = 0; t < 8; ++t)
#pragma unroll
    for (int r = 0; r < 8; ++r) {
      int m = q0 + r + 8 * half;
      ob[(((size_t)b * NQL + m) * HH + h) * HDI + 16 * t + r16] =
          (__bf16)(oacc[t][r] * lrow[r]);
    }
}

// ---------------- launch ---------------------------------------------------

extern "C" void kernel_launch(void* const* d_in, const int* in_sizes, int n_in,
                              void* d_out, int out_size, void* d_ws, size_t ws_size,
                              hipStream_t stream) {
  const float* x          = (const float*)d_in[0];
  const float* k          = (const float*)d_in[1];
  const float* v          = (const float*)d_in[2];
  const int*   tier_lbl   = (const int*)d_in[3];
  const unsigned char* pm = (const unsigned char*)d_in[4];
  const float* q_w        = (const float*)d_in[5];
  const float* q_b        = (const float*)d_in[6];
  const float* out_w      = (const float*)d_in[7];
  const float* out_b      = (const float*)d_in[8];
  const float* tier_bias  = (const float*)d_in[9];

  const size_t nBig = (size_t)BB * HH * NKL * HDI;  // 8,388,608
  const size_t nW   = (size_t)DD * DD;              // 4,194,304
  __bf16* q_ws  = (__bf16*)d_ws;
  __bf16* k_ws  = q_ws  + nBig;
  __bf16* vt_ws = k_ws  + nBig;
  __bf16* o_ws  = vt_ws + nBig;
  __bf16* x_ws  = o_ws  + nBig;
  __bf16* wq_ws = x_ws  + nBig;
  __bf16* wo_ws = wq_ws + nW;

  // 1) fp32 -> bf16 prep (x, q_w, out_w, K, V^T)
  conv_bf16_kernel<<<dim3((unsigned)(nBig / 256)), dim3(256), 0, stream>>>(x, x_ws);
  conv_bf16_kernel<<<dim3((unsigned)(nW / 256)),   dim3(256), 0, stream>>>(q_w, wq_ws);
  conv_bf16_kernel<<<dim3((unsigned)(nW / 256)),   dim3(256), 0, stream>>>(out_w, wo_ws);
  kv_prep_kernel<<<dim3((unsigned)(nBig / 256)),   dim3(256), 0, stream>>>(k, v, k_ws, vt_ws);

  // 2) Q projection: bf16 GEMM -> bf16 [B,H,NQ,HD]
  gemm_nt_kernel<<<dim3(BB * NQL / 32, DD / 128), dim3(128), 0, stream>>>(
      x_ws, wq_ws, q_b, (void*)q_ws, DD, 0);

  // 3) Flash attention
  attn_kernel<<<dim3(NQL / 64, HH, BB), dim3(128), 0, stream>>>(
      q_ws, k_ws, vt_ws, tier_lbl, pm, tier_bias, o_ws);

  // 4) Output projection: bf16 GEMM -> fp32 d_out
  gemm_nt_kernel<<<dim3(BB * NQL / 32, DD / 128), dim3(128), 0, stream>>>(
      o_ws, wo_ws, out_b, d_out, DD, 1);
}
